// Qwen3_5VisionAttention_25967372272040
// MI455X (gfx1250) — matrix operand order repaired and checked by reading, not compiled, not run on hardware
//
#include <hip/hip_runtime.h>
#include <hip/hip_bf16.h>

// ---------------------------------------------------------------------------
// Qwen3.5 Vision Attention for MI455X (gfx1250, wave32, WMMA)
//   S=8192, dim=1280, H=16, hd=80 (padded to 96), 8 segments of L=1024
// Pipeline:
//   [0] f32->f16 converts (hidden, qkv_w, proj_w)
//   [1] WMMA GEMM: qkv_h = hs @ qkv_w^T + qkv_b          (f16 out)
//   [2] RoPE + repack: qf/kf [BH,L,96] f16, vt [BH,96,L] f16 (zero padded)
//   [3] Flash attention per (seg,head), WMMA scores + WMMA P*V
//   [4] WMMA GEMM: out = attn @ proj_w^T + proj_b        (f32 out)
// ---------------------------------------------------------------------------

typedef __attribute__((ext_vector_type(16))) _Float16 v16h;
typedef __attribute__((ext_vector_type(8)))  _Float16 v8h;
typedef __attribute__((ext_vector_type(8)))  float    v8f;

// --- fragment loaders matching CDNA5 16-bit WMMA VGPR layouts ---------------
// A (16x32, MxK), row-major source, K contiguous.
// lane: M = lane&15 ; K-base = 8*(lane>>4); halves 0-7 -> K=kb+0..7,
// halves 8-15 -> K=kb+16..23  (two 16-byte loads).
__device__ __forceinline__ v16h load_a_frag(const _Float16* src, int ld) {
  const int lane = threadIdx.x & 31;
  const int m = lane & 15, hi = lane >> 4;
  const _Float16* p = src + (size_t)m * ld + hi * 8;
  v8h lo = *(const v8h*)(p);
  v8h hb = *(const v8h*)(p + 16);
  v16h r;
#pragma unroll
  for (int i = 0; i < 8; ++i) { r[i] = lo[i]; r[i + 8] = hb[i]; }
  return r;
}

// B (32x16, KxN), with B(k,n) = src[n*ld + k] (i.e. source is [N,K] row-major,
// K contiguous). lane: N = lane&15 ; K-base = 8*(lane>>4); halves 0-7 -> K=kb+0..7,
// halves 8-15 -> K=kb+16..23  (two 16-byte loads).
__device__ __forceinline__ v16h load_b_frag(const _Float16* src, int ld) {
  const int lane = threadIdx.x & 31;
  const int n = lane & 15, hi = lane >> 4;
  const _Float16* p = src + (size_t)n * ld + hi * 8;
  v8h lo = *(const v8h*)(p);
  v8h hb = *(const v8h*)(p + 16);
  v16h r;
#pragma unroll
  for (int i = 0; i < 8; ++i) { r[i] = lo[i]; r[i + 8] = hb[i]; }
  return r;
}

__device__ __forceinline__ v8f wmma16(v16h a, v16h b, v8f c) {
  // v_wmma_f32_16x16x32_f16  (neg_a, A, neg_b, B, c_mod, C, reuse_a, reuse_b)
  return __builtin_amdgcn_wmma_f32_16x16x32_f16(false, a, false, b,
                                                (short)0, c, false, false);
}

// --- [0] f32 -> f16 ---------------------------------------------------------
__global__ void f32_to_f16_kernel(const float* __restrict__ in,
                                  _Float16* __restrict__ out, int n) {
  int i = blockIdx.x * blockDim.x + threadIdx.x;
  if (i < n) out[i] = (_Float16)in[i];
}

// --- [1]/[4] WMMA GEMM: out[M,N] = A[M,K] @ Bw[N,K]^T + bias ---------------
// block = 256 threads = 8 waves (2 M x 4 N); block tile 128x128;
// wave tile 64x32 (4x2 16x16 accumulators); K step 32.
template <bool OUT16>
__global__ void __launch_bounds__(256)
gemm_bias_kernel(const _Float16* __restrict__ A, const _Float16* __restrict__ Bw,
                 const float* __restrict__ bias, void* __restrict__ outp,
                 int M, int N, int K) {
  const int w = threadIdx.x >> 5;
  const int lane = threadIdx.x & 31;
  const int wm = w & 1, wn = w >> 1;
  const int m0 = blockIdx.x * 128 + wm * 64;
  const int n0 = blockIdx.y * 128 + wn * 32;

  v8f acc[4][2] = {};
  for (int k0 = 0; k0 < K; k0 += 32) {
    v16h a[4], b[2];
#pragma unroll
    for (int i = 0; i < 4; ++i)
      a[i] = load_a_frag(A + (size_t)(m0 + 16 * i) * K + k0, K);
#pragma unroll
    for (int j = 0; j < 2; ++j)
      b[j] = load_b_frag(Bw + (size_t)(n0 + 16 * j) * K + k0, K);
#pragma unroll
    for (int i = 0; i < 4; ++i)
#pragma unroll
      for (int j = 0; j < 2; ++j)
        acc[i][j] = wmma16(a[i], b[j], acc[i][j]);
  }

  const int col = lane & 15, hi = lane >> 4;
#pragma unroll
  for (int i = 0; i < 4; ++i) {
#pragma unroll
    for (int j = 0; j < 2; ++j) {
      const int c = n0 + 16 * j + col;
      const float bv = bias[c];
#pragma unroll
      for (int r = 0; r < 8; ++r) {
        const int row = m0 + 16 * i + r + 8 * hi;  // C layout: VGPR r <-> M=r(+8)
        const float v = acc[i][j][r] + bv;
        if (OUT16) ((_Float16*)outp)[(size_t)row * N + c] = (_Float16)v;
        else       ((float*)outp)[(size_t)row * N + c] = v;
      }
    }
  }
}

// --- [2] RoPE + repack ------------------------------------------------------
// qkv_h: f16 [S, 3*1280];  outputs qf/kf: [B*H, 1024, 96], vt: [B*H, 96, 1024]
__global__ void rope_pack_kernel(const _Float16* __restrict__ qkv,
                                 const float* __restrict__ cosb,
                                 const float* __restrict__ sinb,
                                 _Float16* __restrict__ qf,
                                 _Float16* __restrict__ kf,
                                 _Float16* __restrict__ vt) {
  const int total = 8192 * 16 * 96;
  int idx = blockIdx.x * blockDim.x + threadIdx.x;
  if (idx >= total) return;
  const int dp = idx % 96;
  int t = idx / 96;
  const int h = t % 16;
  const int s = t / 16;
  const int b = s >> 10, l = s & 1023;
  const size_t bh = (size_t)(b * 16 + h);

  _Float16 qv = (_Float16)0.f, kv = (_Float16)0.f, vv = (_Float16)0.f;
  if (dp < 80) {
    const size_t base = (size_t)s * 3840 + h * 80;
    const float q = (float)qkv[base + dp];
    const float k = (float)qkv[base + 1280 + dp];
    const float v = (float)qkv[base + 2560 + dp];
    const int dr = (dp < 40) ? dp + 40 : dp - 40;
    const float sg = (dp < 40) ? -1.f : 1.f;
    const float qr = sg * (float)qkv[base + dr];
    const float kr = sg * (float)qkv[base + 1280 + dr];
    const float c = cosb[(size_t)s * 80 + dp];
    const float sn = sinb[(size_t)s * 80 + dp];
    qv = (_Float16)(q * c + qr * sn);
    kv = (_Float16)(k * c + kr * sn);
    vv = (_Float16)v;
  }
  qf[(bh * 1024 + l) * 96 + dp] = qv;
  kf[(bh * 1024 + l) * 96 + dp] = kv;
  vt[(bh * 96 + dp) * 1024 + l] = vv;
}

// --- [3] Flash attention ----------------------------------------------------
// grid = (L/64, B*H); block = 128 (4 waves); each wave owns 16 query rows.
__global__ void __launch_bounds__(128)
attn_kernel(const _Float16* __restrict__ qf, const _Float16* __restrict__ kf,
            const _Float16* __restrict__ vt, _Float16* __restrict__ aout) {
  const int L = 1024, HDP = 96;
  const int w = threadIdx.x >> 5, lane = threadIdx.x & 31;
  const int col = lane & 15, hi = lane >> 4;
  const int bh = blockIdx.y;
  const int q0 = blockIdx.x * 64 + w * 16;

  const _Float16* qb = qf + (size_t)bh * L * HDP;
  const _Float16* kb = kf + (size_t)bh * L * HDP;
  const _Float16* vb = vt + (size_t)bh * HDP * L;

  __shared__ _Float16 pbuf[4][16 * 32];  // per-wave P-tile bounce (C -> A layout)
  _Float16* pw = pbuf[w];

  v16h qa[3];
#pragma unroll
  for (int t = 0; t < 3; ++t)
    qa[t] = load_a_frag(qb + (size_t)q0 * HDP + t * 32, HDP);

  v8f acc[6] = {};
  float mrow[8], lrow[8];
#pragma unroll
  for (int r = 0; r < 8; ++r) { mrow[r] = -1e30f; lrow[r] = 0.f; }
  const float scale = 0.11180339887498949f;  // 1/sqrt(80)

  for (int j = 0; j < L; j += 32) {
    // scores: two 16x16 tiles, K = 96 in 3 WMMA steps
    v8f s0 = {}, s1 = {};
#pragma unroll
    for (int t = 0; t < 3; ++t) {
      v16h b0 = load_b_frag(kb + (size_t)j * HDP + t * 32, HDP);
      v16h b1 = load_b_frag(kb + (size_t)(j + 16) * HDP + t * 32, HDP);
      s0 = wmma16(qa[t], b0, s0);
      s1 = wmma16(qa[t], b1, s1);
    }
    // online softmax update; row = r + 8*hi, cols vary over 16-lane half
#pragma unroll
    for (int r = 0; r < 8; ++r) {
      float a0 = s0[r] * scale, a1 = s1[r] * scale;
      float mx = fmaxf(a0, a1);
#pragma unroll
      for (int off = 1; off < 16; off <<= 1)
        mx = fmaxf(mx, __shfl_xor(mx, off, 32));
      const float mnew = fmaxf(mrow[r], mx);
      const float p0 = __expf(a0 - mnew);
      const float p1 = __expf(a1 - mnew);
      const float alpha = __expf(mrow[r] - mnew);
      float rs = p0 + p1;
#pragma unroll
      for (int off = 1; off < 16; off <<= 1) rs += __shfl_xor(rs, off, 32);
      lrow[r] = lrow[r] * alpha + rs;
      mrow[r] = mnew;
      const int rr = r + 8 * hi;
      pw[rr * 32 + col] = (_Float16)p0;
      pw[rr * 32 + 16 + col] = (_Float16)p1;
#pragma unroll
      for (int t = 0; t < 6; ++t) acc[t][r] *= alpha;
    }
    // P (16x32 f16) @ V (32 x 96): vt is [96, L] so B-frags are K-contiguous
    v16h pa = load_a_frag(pw, 32);
#pragma unroll
    for (int t = 0; t < 6; ++t) {
      v16h bv = load_b_frag(vb + (size_t)t * 16 * L + j, L);
      acc[t] = wmma16(pa, bv, acc[t]);
    }
  }

  const int b = bh >> 4, h = bh & 15;
#pragma unroll
  for (int r = 0; r < 8; ++r) {
    const float invl = 1.0f / lrow[r];
    const int row = b * 1024 + q0 + r + 8 * hi;
    const size_t base = (size_t)row * 1280 + h * 80;
#pragma unroll
    for (int t = 0; t < 5; ++t)  // t=5 is the hd padding (80..95): discard
      aout[base + t * 16 + col] = (_Float16)(acc[t][r] * invl);
  }
}

// ---------------------------------------------------------------------------
extern "C" void kernel_launch(void* const* d_in, const int* in_sizes, int n_in,
                              void* d_out, int out_size, void* d_ws, size_t ws_size,
                              hipStream_t stream) {
  (void)in_sizes; (void)n_in; (void)out_size; (void)ws_size;
  const float* hs     = (const float*)d_in[0];
  // d_in[1] = cu_seqlens (uniform 1024-length segments assumed, as reference)
  const float* cosb   = (const float*)d_in[2];
  const float* sinb   = (const float*)d_in[3];
  const float* qkv_w  = (const float*)d_in[4];
  const float* qkv_b  = (const float*)d_in[5];
  const float* proj_w = (const float*)d_in[6];
  const float* proj_b = (const float*)d_in[7];
  float* out = (float*)d_out;

  const int S = 8192, DIM = 1280, N3 = 3840, BH = 128, HDP = 96, L = 1024;

  // workspace layout (all offsets multiple of 256 bytes)
  char* ws = (char*)d_ws;
  _Float16* hs_h  = (_Float16*)(ws);                       // 8192*1280
  _Float16* w1_h  = (_Float16*)(ws + 20971520);            // 3840*1280
  _Float16* wp_h  = (_Float16*)(ws + 30801920);            // 1280*1280
  _Float16* qkv_h = (_Float16*)(ws + 34078720);            // 8192*3840
  _Float16* qf    = (_Float16*)(ws + 96993280);            // 128*1024*96
  _Float16* kf    = (_Float16*)(ws + 122159104);           // 128*1024*96
  _Float16* vt    = (_Float16*)(ws + 147324928);           // 128*96*1024
  _Float16* aout  = (_Float16*)(ws + 172490752);           // 8192*1280

  // [0] convert inputs to f16
  {
    int n = S * DIM;
    f32_to_f16_kernel<<<(n + 255) / 256, 256, 0, stream>>>(hs, hs_h, n);
    n = N3 * DIM;
    f32_to_f16_kernel<<<(n + 255) / 256, 256, 0, stream>>>(qkv_w, w1_h, n);
    n = DIM * DIM;
    f32_to_f16_kernel<<<(n + 255) / 256, 256, 0, stream>>>(proj_w, wp_h, n);
  }

  // [1] QKV GEMM (f16 out, bias fused)
  gemm_bias_kernel<true><<<dim3(S / 128, N3 / 128), 256, 0, stream>>>(
      hs_h, w1_h, qkv_b, (void*)qkv_h, S, N3, DIM);

  // [2] RoPE + repack into WMMA-friendly padded layouts
  {
    int n = S * 16 * HDP;
    rope_pack_kernel<<<(n + 255) / 256, 256, 0, stream>>>(qkv_h, cosb, sinb,
                                                          qf, kf, vt);
  }

  // [3] flash attention: grid (L/64 q-tiles, B*H)
  attn_kernel<<<dim3(L / 64, BH), 128, 0, stream>>>(qf, kf, vt, aout);

  // [4] output projection (f32 out, bias fused)
  gemm_bias_kernel<false><<<dim3(S / 128, DIM / 128), 256, 0, stream>>>(
      aout, wp_h, proj_b, (void*)out, S, DIM, DIM);
}
